// CrystalGraphConvNet_10170482557306
// MI455X (gfx1250) — matrix operand order repaired and checked by reading, not compiled, not run on hardware
//
#include <hip/hip_runtime.h>

// ---------------- problem constants ----------------
#define NATOMS 65536
#define MN     12
#define ORIGF  92
#define NBRF   41
#define NBRP   48    // edge features padded to 48 bf16 (96B rows, 16B aligned)
#define AFEA   64
#define HFEA   128
#define N0C    1024
#define KEMB   96    // 92 padded to 3*32

typedef __attribute__((ext_vector_type(16))) __bf16 v16bf;
typedef __attribute__((ext_vector_type(8)))  float  v8f;
typedef __attribute__((ext_vector_type(8)))  unsigned int v8u;

union Frag {
  v16bf bf;
  v8u   u8;
  unsigned w[8];
  uint4 q[2];
};

__device__ __forceinline__ unsigned short f2bf(float f) {
  unsigned u = __float_as_uint(f);
  unsigned r = 0x7FFFu + ((u >> 16) & 1u);     // round-to-nearest-even
  return (unsigned short)((u + r) >> 16);
}
__device__ __forceinline__ float softplusf(float x) {
  return (x > 20.f) ? x : log1pf(__expf(x));
}
__device__ __forceinline__ float sigmoidf(float x) {
  return 1.f / (1.f + __expf(-x));
}

// ---------------- prep kernels ----------------
__global__ void zero_f32(float* p, int n) {
  int i = blockIdx.x * blockDim.x + threadIdx.x;
  if (i < n) p[i] = 0.f;
}

// pad atom_fea (N,92) f32 -> (N,96) bf16 bits
__global__ void prep_apad(const float* __restrict__ src, unsigned short* __restrict__ dst) {
  int i = blockIdx.x * blockDim.x + threadIdx.x;
  if (i >= NATOMS * KEMB) return;
  int r = i / KEMB, k = i - r * KEMB;
  float v = (k < ORIGF) ? src[r * ORIGF + k] : 0.f;
  dst[i] = f2bf(v);
}

// pad nbr_fea (N*M,41) f32 -> (N*M,48) bf16 bits
__global__ void prep_nbrf(const float* __restrict__ src, unsigned short* __restrict__ dst) {
  int i = blockIdx.x * blockDim.x + threadIdx.x;
  if (i >= NATOMS * MN * NBRP) return;
  int r = i / NBRP, k = i - r * NBRP;
  float v = (k < NBRF) ? src[(size_t)r * NBRF + k] : 0.f;
  dst[i] = f2bf(v);
}

// pack emb_w (92,64) into WMMA B fragments: [cg=4][t=3][lane=32][v=8] u32
__global__ void prep_wemb(const float* __restrict__ src, unsigned* __restrict__ dst) {
  int i = blockIdx.x * blockDim.x + threadIdx.x;
  if (i >= 4 * 3 * 32 * 8) return;
  int v = i & 7;
  int lane = (i >> 3) & 31;
  int rest = i >> 8;            // cg*3 + t
  int t = rest % 3, cg = rest / 3;
  int col = cg * 16 + (lane & 15);
  int k0 = t * 32 + ((lane >> 4) * 8) + 2 * v + ((v >> 2) * 8);
  float f0 = (k0 < ORIGF)     ? src[k0 * AFEA + col]       : 0.f;
  float f1 = (k0 + 1 < ORIGF) ? src[(k0 + 1) * AFEA + col] : 0.f;
  dst[i] = (unsigned)f2bf(f0) | ((unsigned)f2bf(f1) << 16);
}

// pack conv_fc_w (3,169,128) into [layer][cg=8][t=6][lane=32][v=8] u32 (K padded to 192)
__global__ void prep_wconv(const float* __restrict__ src, unsigned* __restrict__ dst) {
  int i = blockIdx.x * blockDim.x + threadIdx.x;
  if (i >= 3 * 8 * 6 * 32 * 8) return;
  int v = i & 7;
  int lane = (i >> 3) & 31;
  int rest = i >> 8;            // ((layer*8+cg)*6+t)
  int t = rest % 6;
  int cg = (rest / 6) & 7;
  int layer = rest / 48;
  int col = cg * 16 + (lane & 15);
  int k0 = t * 32 + ((lane >> 4) * 8) + 2 * v + ((v >> 2) * 8);
  float f0 = (k0 < 169)     ? src[((size_t)layer * 169 + k0) * 128 + col]     : 0.f;
  float f1 = (k0 + 1 < 169) ? src[((size_t)layer * 169 + k0 + 1) * 128 + col] : 0.f;
  dst[i] = (unsigned)f2bf(f0) | ((unsigned)f2bf(f1) << 16);
}

// ---------------- embedding GEMM: (65536x96)bf16 @ (96x64)bf16 -> afea f32 + bf16 mirror ----
__global__ __launch_bounds__(256)
void emb_gemm(const unsigned short* __restrict__ Apad, const unsigned* __restrict__ Bpk,
              const float* __restrict__ bias, float* __restrict__ afea,
              unsigned short* __restrict__ afeabf) {
  int lane = threadIdx.x & 31;
  int wave = threadIdx.x >> 5;
  int r0 = (blockIdx.x * 8 + wave) * 16;        // 16 rows per wave
  int m = lane & 15, kh = lane >> 4;

  v8f acc[4];
#pragma unroll
  for (int c = 0; c < 4; ++c)
#pragma unroll
    for (int j = 0; j < 8; ++j) acc[c][j] = 0.f;

  const uint4* rp = (const uint4*)(Apad + (size_t)(r0 + m) * KEMB);
#pragma unroll
  for (int t = 0; t < 3; ++t) {
    Frag a;
    a.q[0] = rp[t * 4 + kh];          // K group 0: words t*16 + kh*4 + 0..3
    a.q[1] = rp[t * 4 + 2 + kh];      // K group 1: words t*16 + 8 + kh*4 + 0..3
#pragma unroll
    for (int cg = 0; cg < 4; ++cg) {
      Frag b;
      b.u8 = *(const v8u*)(Bpk + ((size_t)(cg * 3 + t) * 32 + lane) * 8);
      acc[cg] = __builtin_amdgcn_wmma_f32_16x16x32_bf16(
          false, a.bf, false, b.bf, (short)0, acc[cg], false, false);
    }
  }
#pragma unroll
  for (int cg = 0; cg < 4; ++cg) {
    int col = cg * 16 + m;
    float bv = bias[col];
#pragma unroll
    for (int p = 0; p < 8; ++p) {
      int row = r0 + p + kh * 8;
      float v = acc[cg][p] + bv;
      afea[(size_t)row * AFEA + col] = v;
      afeabf[(size_t)row * AFEA + col] = f2bf(v);
    }
  }
}

// ---------------- conv GEMM (both passes) ----------------
// Each block owns 16 atoms = 192 rows; 4 waves x 3 tiles of 16 rows.
// A-fragments loaded directly from global bf16 buffers as b128 pairs:
//   K layout per row: [0,64)=self afea | [64,128)=neighbor afea | [128,176)=edge | [176,192)=0
// mode 0: accumulate per-column sum/sumsq of gated (=acc+bias) into gsum/gsumsq[128]
// mode 1: y = a1*acc + bb1 (bias+BN folded); sigmoid(filt)*softplus(core); sum over
//         neighbors into nsOut (N,64); BN2 stats into gsum2/gsumsq2[64]
__global__ __launch_bounds__(128)
void conv_gemm(const unsigned short* __restrict__ afeabf,
               const unsigned short* __restrict__ nbrfbf,
               const int* __restrict__ nbrIdx, const unsigned* __restrict__ Bpk,
               const float* __restrict__ bias,
               float* __restrict__ gsum, float* __restrict__ gsumsq,
               const float* __restrict__ a1, const float* __restrict__ bb1,
               float* __restrict__ nsOut, float* __restrict__ gsum2,
               float* __restrict__ gsumsq2, int mode) {
  __shared__ float sNS[16 * AFEA];              // per-atom neighbor sums
  __shared__ float sS[128], sS2[128];           // per-column stats partials

  int tid = threadIdx.x, lane = tid & 31, wave = tid >> 5;
  int m = lane & 15, kh = lane >> 4;
  int atom0 = blockIdx.x * 16;

  for (int i = tid; i < 16 * AFEA; i += 128) sNS[i] = 0.f;
  if (tid < 128) { sS[tid] = 0.f; sS2[tid] = 0.f; }
  __syncthreads();

  for (int tile = 0; tile < 3; ++tile) {
    int rib = wave * 48 + tile * 16 + m;        // this lane's A row (row-in-block)
    int atom = atom0 + rib / MN;
    int mm = rib % MN;
    int nb = nbrIdx[atom * MN + mm];
    const uint4* ps = (const uint4*)(afeabf + (size_t)atom * AFEA);
    const uint4* pn = (const uint4*)(afeabf + (size_t)nb * AFEA);
    const uint4* pe = (const uint4*)(nbrfbf + ((size_t)atom * MN + mm) * NBRP);

    v8f acc[8];
#pragma unroll
    for (int c = 0; c < 8; ++c)
#pragma unroll
      for (int j = 0; j < 8; ++j) acc[c][j] = 0.f;

#pragma unroll
    for (int t = 0; t < 6; ++t) {
      Frag a;
      if (t < 2) {                 // self segment, K in [0,64)
        a.q[0] = ps[t * 4 + kh];
        a.q[1] = ps[t * 4 + 2 + kh];
      } else if (t < 4) {          // neighbor segment, K in [64,128)
        a.q[0] = pn[(t - 2) * 4 + kh];
        a.q[1] = pn[(t - 2) * 4 + 2 + kh];
      } else if (t == 4) {         // edge segment, K in [128,160)
        a.q[0] = pe[kh];
        a.q[1] = pe[2 + kh];
      } else {                     // K in [160,176) valid edge, [176,192)=0
        a.q[0] = pe[4 + kh];
        a.q[1] = make_uint4(0u, 0u, 0u, 0u);
      }
#pragma unroll
      for (int cg = 0; cg < 8; ++cg) {
        Frag b;
        b.u8 = *(const v8u*)(Bpk + ((size_t)(cg * 6 + t) * 32 + lane) * 8);
        acc[cg] = __builtin_amdgcn_wmma_f32_16x16x32_bf16(
            false, a.bf, false, b.bf, (short)0, acc[cg], false, false);
      }
    }

    if (mode == 0) {
#pragma unroll
      for (int cg = 0; cg < 8; ++cg) {
        int col = cg * 16 + m;
        float bv = bias[col];
        float s = 0.f, s2 = 0.f;
#pragma unroll
        for (int p = 0; p < 8; ++p) {
          float x = acc[cg][p] + bv;
          s += x; s2 += x * x;
        }
        atomicAdd(&sS[col], s);
        atomicAdd(&sS2[col], s2);
      }
    } else {
#pragma unroll
      for (int cg = 0; cg < 4; ++cg) {
        int colf = cg * 16 + m, colc = colf + 64;
        float af = a1[colf], bf_ = bb1[colf];
        float ac = a1[colc], bc_ = bb1[colc];
#pragma unroll
        for (int p = 0; p < 8; ++p) {
          float yf = af * acc[cg][p] + bf_;
          float yc = ac * acc[cg + 4][p] + bc_;
          float val = sigmoidf(yf) * softplusf(yc);
          int rowb = wave * 48 + tile * 16 + p + kh * 8;   // C-layout row
          int la = rowb / MN;
          atomicAdd(&sNS[la * AFEA + colf], val);
        }
      }
    }
  }
  __syncthreads();

  if (mode == 0) {
    if (tid < 128) {
      atomicAdd(&gsum[tid], sS[tid]);
      atomicAdd(&gsumsq[tid], sS2[tid]);
    }
  } else {
    for (int e = tid; e < 16 * AFEA; e += 128) {
      int la = e >> 6, c = e & 63;
      nsOut[(size_t)(atom0 + la) * AFEA + c] = sNS[e];
    }
    if (tid < 64) {
      float s = 0.f, s2 = 0.f;
#pragma unroll
      for (int la = 0; la < 16; ++la) {
        float v = sNS[la * AFEA + tid];
        s += v; s2 += v * v;
      }
      atomicAdd(&gsum2[tid], s);
      atomicAdd(&gsumsq2[tid], s2);
    }
  }
}

// BN1 finalize: fold bias into affine. y = a1*acc + bb1 for raw accumulator.
__global__ void finalize1(const float* gsum, const float* gsumsq, const float* g,
                          const float* bnb, const float* bias,
                          float* a1, float* bb1, float R) {
  int c = threadIdx.x;   // 128
  float mu = gsum[c] / R;                    // mean of (acc+bias)
  float var = gsumsq[c] / R - mu * mu;
  float a = g[c] * rsqrtf(fmaxf(var, 0.f) + 1e-5f);
  a1[c] = a;
  bb1[c] = bnb[c] + a * (bias[c] - mu);
}

__global__ void finalize2(const float* gsum, const float* gsumsq, const float* g,
                          const float* bnb, float* a2, float* bb2, float R) {
  int c = threadIdx.x;   // 64
  float mu = gsum[c] / R;
  float var = gsumsq[c] / R - mu * mu;
  float a = g[c] * rsqrtf(fmaxf(var, 0.f) + 1e-5f);
  a2[c] = a;
  bb2[c] = bnb[c] - a * mu;
}

// afea <- softplus(afea + BN2(ns)) + afea   (identity == afea at layer entry)
__global__ void update_afea(float* __restrict__ afea, const float* __restrict__ ns,
                            const float* __restrict__ a2, const float* __restrict__ bb2,
                            unsigned short* __restrict__ afeabf) {
  int i = blockIdx.x * blockDim.x + threadIdx.x;
  if (i >= NATOMS * AFEA) return;
  int c = i & 63;
  float y = a2[c] * ns[i] + bb2[c];
  float x = afea[i];
  float v = softplusf(x + y) + x;
  afea[i] = v;
  afeabf[i] = f2bf(v);
}

// segment sum over crystals
__global__ void pool_sum(const float* __restrict__ afea, const int* __restrict__ cidx,
                         float* __restrict__ crys, float* __restrict__ cnt) {
  int i = blockIdx.x * blockDim.x + threadIdx.x;
  if (i >= NATOMS * AFEA) return;
  int atom = i >> 6, c = i & 63;
  int cid = cidx[atom];
  atomicAdd(&crys[cid * AFEA + c], afea[i]);
  if (c == 0) atomicAdd(&cnt[cid], 1.f);
}

// head: mean -> softplus -> fc1 -> softplus (crys out) -> out1 -> softplus -> out2
__global__ __launch_bounds__(128)
void head(const float* __restrict__ crysSum, const float* __restrict__ cnt,
          const float* __restrict__ fc1w, const float* __restrict__ fc1b,
          const float* __restrict__ o1w, const float* __restrict__ o1b,
          const float* __restrict__ o2w, const float* __restrict__ o2b,
          float* __restrict__ out) {
  __shared__ float p[64], h1[128], h2[64];
  int cr = blockIdx.x, tid = threadIdx.x;
  if (tid < 64) {
    float cc = fmaxf(cnt[cr], 1.f);
    p[tid] = softplusf(crysSum[cr * AFEA + tid] / cc);
  }
  __syncthreads();
  {
    float s = fc1b[tid];
    for (int k = 0; k < 64; ++k) s += p[k] * fc1w[k * HFEA + tid];
    float v = softplusf(s);
    h1[tid] = v;
    out[N0C + (size_t)cr * HFEA + tid] = v;   // crys output
  }
  __syncthreads();
  if (tid < 64) {
    float s = o1b[tid];
    for (int k = 0; k < 128; ++k) s += h1[k] * o1w[k * 64 + tid];
    h2[tid] = softplusf(s);
  }
  __syncthreads();
  if (tid == 0) {
    float s = o2b[0];
    for (int k = 0; k < 64; ++k) s += h2[k] * o2w[k];
    out[cr] = s;
  }
}

// ---------------- workspace layout ----------------
static constexpr size_t OFF_AFEA = 0;
static constexpr size_t SZ_AFEA  = (size_t)NATOMS * AFEA * 4;
static constexpr size_t OFF_NS   = OFF_AFEA + SZ_AFEA;
static constexpr size_t SZ_NS    = (size_t)NATOMS * AFEA * 4;
static constexpr size_t OFF_ABF  = OFF_NS + SZ_NS;
static constexpr size_t SZ_ABF   = (size_t)NATOMS * AFEA * 2;
static constexpr size_t OFF_NBF  = OFF_ABF + SZ_ABF;
static constexpr size_t SZ_NBF   = (size_t)NATOMS * MN * NBRP * 2;
static constexpr size_t OFF_APAD = OFF_NBF + SZ_NBF;
static constexpr size_t SZ_APAD  = (size_t)NATOMS * KEMB * 2;
static constexpr size_t OFF_BEMB = OFF_APAD + SZ_APAD;
static constexpr size_t SZ_BEMB  = 4 * 3 * 32 * 8 * 4;
static constexpr size_t OFF_BCONV = OFF_BEMB + SZ_BEMB;
static constexpr size_t SZ_BCONV  = (size_t)3 * 8 * 6 * 32 * 8 * 4;
static constexpr size_t OFF_ST   = OFF_BCONV + SZ_BCONV;
// floats: [0,128)=sum1 [128,256)=sq1 [256,384)=a1 [384,512)=bb1
//         [512,576)=sum2 [576,640)=sq2 [640,704)=a2 [704,768)=bb2
static constexpr size_t SZ_ST    = 768 * 4;
static constexpr size_t OFF_CRYS = OFF_ST + SZ_ST;
static constexpr size_t SZ_CRYS  = (size_t)N0C * AFEA * 4;
static constexpr size_t OFF_CNT  = OFF_CRYS + SZ_CRYS;

extern "C" void kernel_launch(void* const* d_in, const int* in_sizes, int n_in,
                              void* d_out, int out_size, void* d_ws, size_t ws_size,
                              hipStream_t stream) {
  (void)in_sizes; (void)n_in; (void)out_size; (void)ws_size;
  const float* atom_fea  = (const float*)d_in[0];
  const float* nbr_fea   = (const float*)d_in[1];
  const int*   nbr_idx   = (const int*)d_in[2];
  const int*   cidx      = (const int*)d_in[3];
  const float* emb_w     = (const float*)d_in[4];
  const float* emb_b     = (const float*)d_in[5];
  const float* conv_fc_w = (const float*)d_in[6];
  const float* conv_fc_b = (const float*)d_in[7];
  const float* bn1g      = (const float*)d_in[8];
  const float* bn1b      = (const float*)d_in[9];
  const float* bn2g      = (const float*)d_in[10];
  const float* bn2b      = (const float*)d_in[11];
  const float* fc1w      = (const float*)d_in[12];
  const float* fc1b      = (const float*)d_in[13];
  const float* o1w       = (const float*)d_in[14];
  const float* o1b       = (const float*)d_in[15];
  const float* o2w       = (const float*)d_in[16];
  const float* o2b       = (const float*)d_in[17];
  float* out = (float*)d_out;

  char* w = (char*)d_ws;
  float*          afea  = (float*)(w + OFF_AFEA);
  float*          ns    = (float*)(w + OFF_NS);
  unsigned short* abf   = (unsigned short*)(w + OFF_ABF);
  unsigned short* nbf   = (unsigned short*)(w + OFF_NBF);
  unsigned short* apad  = (unsigned short*)(w + OFF_APAD);
  unsigned*       bemb  = (unsigned*)(w + OFF_BEMB);
  unsigned*       bconv = (unsigned*)(w + OFF_BCONV);
  float*          st    = (float*)(w + OFF_ST);
  float*          crysb = (float*)(w + OFF_CRYS);
  float*          cnt   = (float*)(w + OFF_CNT);

  // ---- prep (bf16 compression keeps the whole GEMM working set L2-resident) ----
  prep_apad<<<(NATOMS * KEMB + 255) / 256, 256, 0, stream>>>(atom_fea, apad);
  prep_nbrf<<<(NATOMS * MN * NBRP + 255) / 256, 256, 0, stream>>>(nbr_fea, nbf);
  prep_wemb<<<(4 * 3 * 32 * 8 + 255) / 256, 256, 0, stream>>>(emb_w, bemb);
  prep_wconv<<<(3 * 8 * 6 * 32 * 8 + 255) / 256, 256, 0, stream>>>(conv_fc_w, bconv);

  // ---- embedding GEMM ----
  emb_gemm<<<NATOMS / 128, 256, 0, stream>>>(apad, bemb, emb_b, afea, abf);

  // ---- conv layers ----
  for (int layer = 0; layer < 3; ++layer) {
    const unsigned* bl = bconv + (size_t)layer * 8 * 6 * 32 * 8;
    const float* bias = conv_fc_b + layer * 128;

    zero_f32<<<1, 256, 0, stream>>>(st, 256);
    conv_gemm<<<NATOMS / 16, 128, 0, stream>>>(abf, nbf, nbr_idx, bl, bias,
                                               st, st + 128, st + 256, st + 384,
                                               ns, st + 512, st + 576, 0);
    finalize1<<<1, 128, 0, stream>>>(st, st + 128, bn1g + layer * 128,
                                     bn1b + layer * 128, bias,
                                     st + 256, st + 384, (float)(NATOMS * MN));
    zero_f32<<<1, 128, 0, stream>>>(st + 512, 128);
    conv_gemm<<<NATOMS / 16, 128, 0, stream>>>(abf, nbf, nbr_idx, bl, bias,
                                               st, st + 128, st + 256, st + 384,
                                               ns, st + 512, st + 576, 1);
    finalize2<<<1, 64, 0, stream>>>(st + 512, st + 576, bn2g + layer * 64,
                                    bn2b + layer * 64, st + 640, st + 704,
                                    (float)NATOMS);
    update_afea<<<(NATOMS * AFEA + 255) / 256, 256, 0, stream>>>(afea, ns, st + 640,
                                                                 st + 704, abf);
  }

  // ---- pool + head ----
  zero_f32<<<(N0C * AFEA + N0C + 255) / 256, 256, 0, stream>>>(crysb, N0C * AFEA + N0C);
  pool_sum<<<(NATOMS * AFEA + 255) / 256, 256, 0, stream>>>(afea, cidx, crysb, cnt);
  head<<<N0C, 128, 0, stream>>>(crysb, cnt, fc1w, fc1b, o1w, o1b, o2w, o2b, out);
}
